// SmallGraphormer_38465727103220
// MI455X (gfx1250) — matrix operand order repaired and checked by reading, not compiled, not run on hardware
//
#include <hip/hip_runtime.h>

// ---------------------------------------------------------------------------
// Graphormer forward for MI455X (gfx1250, wave32, WMMA bf16 16x16x32)
// B=16 N=512 F=512 H=16 dh=32 HID=2048 L=6
// Data movement: TDM tensor_load_to_lds (GEMM tiles, double-buffered) and
// global_load_async_to_lds_b128 (attention K tiles); compute: v_wmma bf16.
// ---------------------------------------------------------------------------
#define GB   16
#define GN   512
#define GF   512
#define GH   16
#define GHID 2048
#define GL   6
#define GM   (GB * GN)   // 8192 rows of the flattened activation matrix

typedef unsigned short ushort_t;
typedef unsigned int   uint_t;

typedef __bf16 bf16x16 __attribute__((ext_vector_type(16)));
typedef float  f32x8   __attribute__((ext_vector_type(8)));
typedef uint_t u32x4   __attribute__((ext_vector_type(4)));
typedef uint_t u32x8   __attribute__((ext_vector_type(8)));

union FragB16 {
    bf16x16 v;
    ushort_t u[16];
    u32x4    q[2];
};

__device__ __forceinline__ ushort_t f2bf(float f) {
    uint_t x = __float_as_uint(f);
    uint_t r = (x + 0x7FFFu + ((x >> 16) & 1u)) >> 16;  // RNE
    return (ushort_t)r;
}

// ---- CDNA5 async global->LDS (ASYNCcnt) -----------------------------------
__device__ __forceinline__ void async_copy_b128(uint_t lds_byte, const void* g) {
    asm volatile("global_load_async_to_lds_b128 %0, %1, off"
                 :: "v"(lds_byte), "v"((unsigned long long)g)
                 : "memory");
}
__device__ __forceinline__ void async_wait() {
    asm volatile("s_wait_asynccnt 0x0" ::: "memory");
}

// ---- CDNA5 Tensor Data Mover: 2-D bf16 tile (rows x 32) -> LDS ------------
// Row = 32 elems (64B), LDS row stride = 80B via pad (16 dwords + 4 dwords).
__device__ __forceinline__ void tdm_load_tile(const void* g, uint_t lds_byte,
                                              int rows, int strideElems) {
    unsigned long long ga = (unsigned long long)g;
    u32x4 g0;
    g0[0] = 1u;                                    // count=1 (valid user D#)
    g0[1] = lds_byte;                              // lds_addr
    g0[2] = (uint_t)ga;                            // global_addr[31:0]
    g0[3] = (uint_t)((ga >> 32) & 0x01FFFFFFull)   // global_addr[56:32]
            | 0x80000000u;                         // type=2 ("image")
    u32x8 g1;
    g1[0] = (1u << 16)      // data_size = 2 bytes
          | (1u << 20)      // pad_enable
          | (3u << 22)      // pad_interval = 16 dwords (64B)
          | (3u << 25);     // pad_amount   = 4 dwords (16B)
    g1[1] = (32u << 16);                  // tensor_dim0 = 32 elems
    g1[2] = ((uint_t)rows << 16);         // tensor_dim1 = rows
    g1[3] = (32u << 16);                  // tile_dim0 = 32
    g1[4] = (uint_t)rows;                 // tile_dim1 = rows
    g1[5] = (uint_t)strideElems;          // tensor_dim0_stride (elems)
    g1[6] = 0u;
    g1[7] = 0u;
    asm volatile("tensor_load_to_lds %0, %1" :: "s"(g0), "s"(g1) : "memory");
}

// ---------------------------------------------------------------------------
// Weight convert + transpose: W (K x Nn f32 row-major) -> Wt (Nn x K bf16)
// ---------------------------------------------------------------------------
__global__ void convt_kernel(const float* __restrict__ W,
                             ushort_t* __restrict__ Wt, int K, int Nn) {
    size_t idx = (size_t)blockIdx.x * blockDim.x + threadIdx.x;
    int k = (int)(idx / Nn);
    int n = (int)(idx % Nn);
    Wt[(size_t)n * K + k] = f2bf(W[idx]);
}

// ---------------------------------------------------------------------------
// Input embedding: x = valid ? nfeats + deg_in_emb[deg] + deg_out_emb[deg] : 0
// ---------------------------------------------------------------------------
__global__ void embed_kernel(const float* __restrict__ nfeats,
                             const int* __restrict__ degrees,
                             const int* __restrict__ num_nodes,
                             const float* __restrict__ din,
                             const float* __restrict__ dout,
                             float* __restrict__ x, ushort_t* __restrict__ xb) {
    size_t idx = (size_t)blockIdx.x * blockDim.x + threadIdx.x;
    int f = (int)(idx & (GF - 1));
    int n = (int)((idx >> 9) & (GN - 1));
    int b = (int)(idx >> 18);
    float v = 0.0f;
    if (n < num_nodes[b]) {
        int dg = degrees[b * GN + n];
        dg = min(max(dg, 0), 100);
        v = nfeats[idx] + din[(size_t)dg * GF + f] + dout[(size_t)dg * GF + f];
    }
    x[idx]  = v;
    xb[idx] = f2bf(v);
}

// ---------------------------------------------------------------------------
// GEMM: C(MxNn,f32) = A(MxK bf16 row-major) @ Bt(NnxK bf16 "W transposed")
// Block: 256 threads (8 waves). Tile 128(M) x 64(N), K staged 32 at a time
// through double-buffered LDS filled by TDM (wave 0 issues 2 descriptors per
// stage; DMA overlaps the WMMA stream; s_wait_tensorcnt before the barrier).
// Dynamic-LDS byte layout: A0 @0 (10240) | B0 @10240 (5120)
//                          A1 @15360     | B1 @25600   -> 30720 B total.
// ---------------------------------------------------------------------------
__global__ __launch_bounds__(256) void gemm_bf16_kernel(
    const ushort_t* __restrict__ A, const ushort_t* __restrict__ Bt,
    const float* __restrict__ bias, float* __restrict__ Cf,
    ushort_t* __restrict__ Cb, int M, int K, int Nn, float alpha, int relu) {
    extern __shared__ __align__(16) ushort_t smemg[];

    const int tid  = threadIdx.x;
    const int lane = tid & 31;
    const int wv   = tid >> 5;
    const int wm   = wv >> 1;        // 0..3 (M direction)
    const int wn   = wv & 1;         // 0..1 (N direction)
    const int lrow = lane & 15;
    const int hi   = lane >> 4;
    const int m0   = blockIdx.y * 128;
    const int n0   = blockIdx.x * 64;

    const uint_t aByte[2] = {0u, 15360u};
    const uint_t bByte[2] = {10240u, 25600u};

    const f32x8 zero = {0, 0, 0, 0, 0, 0, 0, 0};
    f32x8 acc[2][2];
    acc[0][0] = zero; acc[0][1] = zero; acc[1][0] = zero; acc[1][1] = zero;

    auto issue_stage = [&](int k0, int p) {
        if (wv == 0) {
            tdm_load_tile(A + (size_t)m0 * K + k0, aByte[p], 128, K);
            tdm_load_tile(Bt + (size_t)n0 * K + k0, bByte[p], 64, K);
        }
    };
    auto wait_stage = [&]() {
        if (wv == 0) __builtin_amdgcn_s_wait_tensorcnt(0);
    };

    issue_stage(0, 0);
    wait_stage();
    __syncthreads();

    int p = 0;
    for (int k0 = 0; k0 < K; k0 += 32) {
        if (k0 + 32 < K) issue_stage(k0 + 32, p ^ 1);  // DMA next stage

        const ushort_t* sA = smemg + (aByte[p] >> 1);
        const ushort_t* sB = smemg + (bByte[p] >> 1);
        FragB16 af[2], bfr[2];
#pragma unroll
        for (int i = 0; i < 2; ++i) {
            int row = wm * 32 + i * 16 + lrow;
            int oa  = hi ? 8 : 0;
            af[i].q[0] = *(const u32x4*)&sA[row * 40 + oa];
            af[i].q[1] = *(const u32x4*)&sA[row * 40 + oa + 16];
            int col = wn * 32 + i * 16 + lrow;
            int ob  = hi ? 8 : 0;
            bfr[i].q[0] = *(const u32x4*)&sB[col * 40 + ob];
            bfr[i].q[1] = *(const u32x4*)&sB[col * 40 + ob + 16];
        }
#pragma unroll
        for (int i = 0; i < 2; ++i)
#pragma unroll
            for (int j = 0; j < 2; ++j)
                acc[i][j] = __builtin_amdgcn_wmma_f32_16x16x32_bf16(
                    false, af[i].v, false, bfr[j].v, (short)0, acc[i][j],
                    false, false);

        wait_stage();        // this wave's TDM for next stage done
        __syncthreads();     // publish LDS; all waves done reading buf p
        p ^= 1;
    }

    // ---- epilogue: bias, relu, alpha, store f32/bf16 -------------------
#pragma unroll
    for (int i = 0; i < 2; ++i) {
        int gmB = m0 + wm * 32 + i * 16 + (hi ? 8 : 0);
#pragma unroll
        for (int j = 0; j < 2; ++j) {
            int gn   = n0 + wn * 32 + j * 16 + lrow;
            float bb = bias ? bias[gn] : 0.0f;
#pragma unroll
            for (int r = 0; r < 8; ++r) {
                float v = acc[i][j][r] + bb;
                if (relu) v = fmaxf(v, 0.0f);
                v *= alpha;
                size_t o = (size_t)(gmB + r) * Nn + gn;
                if (Cf) Cf[o] = v;
                if (Cb) Cb[o] = f2bf(v);
            }
        }
    }
}

// ---------------------------------------------------------------------------
// Attention: one block per (b, h, 16-query-row tile). 128 threads (4 waves).
//  S = (scaled Q) K^T (WMMA, K tiles via async global->LDS) + bias + mask
//  softmax (two-pass, 1/sum deferred), O = P V (WMMA) -> bf16 out.
// Dynamic-LDS byte layout:
//  sQ @0 (1280) | sS @1280 (33280) | sK @34560 (10240) | sVt @44800 (8704)
//  sRed @53504 (576) | sO @54080 (8192)  -> 62272 B total.
// ---------------------------------------------------------------------------
#define AT_KBYTE 34560u
__global__ __launch_bounds__(128) void attn_kernel(
    const ushort_t* __restrict__ qb, const ushort_t* __restrict__ kb,
    const ushort_t* __restrict__ vb, const int* __restrict__ dist_idx,
    const float* __restrict__ spatial, const int* __restrict__ num_nodes,
    ushort_t* __restrict__ ob) {
    extern __shared__ __align__(16) char smema[];
    ushort_t* sQ   = (ushort_t*)(smema);
    float*    sS   = (float*)(smema + 1280);
    ushort_t* sK   = (ushort_t*)(smema + 34560);
    ushort_t* sVt  = (ushort_t*)(smema + 44800);
    float*    sRed = (float*)(smema + 53504);
    float*    sO   = (float*)(smema + 54080);

    const int tid  = threadIdx.x;
    const int lane = tid & 31;
    const int wv   = tid >> 5;   // 0..3
    const int lrow = lane & 15;
    const int hi   = lane >> 4;

    const int tI = blockIdx.x;
    const int qt = tI & 31;
    const int h  = (tI >> 5) & 15;
    const int b  = tI >> 9;
    const int qi = qt * 16;
    const int nn = num_nodes[b];

    // ---- load Q tile (16x32 bf16) --------------------------------------
    if (tid < 64) {
        int r = tid >> 2, c4 = tid & 3;
        const u32x4* qp =
            (const u32x4*)(qb + ((size_t)(b * GN + qi + r)) * GF + h * 32);
        *(u32x4*)&sQ[r * 40 + c4 * 8] = qp[c4];
    }
    __syncthreads();
    FragB16 qf;
    {
        int oa = hi ? 8 : 0;
        qf.q[0] = *(const u32x4*)&sQ[lrow * 40 + oa];
        qf.q[1] = *(const u32x4*)&sQ[lrow * 40 + oa + 16];
    }

    const f32x8 zero = {0, 0, 0, 0, 0, 0, 0, 0};

    // ---- phase 1: scores + bias + mask ---------------------------------
    for (int kc = 0; kc < GN; kc += 128) {
        __syncthreads();  // previous stage's sK reads complete
        {                 // async DMA: 128 K-rows (32 bf16 each) -> LDS
            const ushort_t* kp =
                kb + ((size_t)(b * GN + kc + tid)) * GF + h * 32;
#pragma unroll
            for (int c4 = 0; c4 < 4; ++c4)
                async_copy_b128(AT_KBYTE + (uint_t)(tid * 80 + c4 * 16),
                                kp + c4 * 8);
        }
        async_wait();
        __syncthreads();
#pragma unroll
        for (int nt = 0; nt < 2; ++nt) {
            int colL = wv * 32 + nt * 16 + lrow;
            FragB16 kf;
            int ob2 = hi ? 8 : 0;
            kf.q[0] = *(const u32x4*)&sK[colL * 40 + ob2];
            kf.q[1] = *(const u32x4*)&sK[colL * 40 + ob2 + 16];
            f32x8 sc = __builtin_amdgcn_wmma_f32_16x16x32_bf16(
                false, qf.v, false, kf.v, (short)0, zero, false, false);
            int j = kc + colL;
            const int* drow = dist_idx + (size_t)b * GN * GN + j;
#pragma unroll
            for (int r = 0; r < 8; ++r) {
                int iloc = r + (hi ? 8 : 0);
                int i    = qi + iloc;
                float v  = sc[r] + spatial[drow[(size_t)i * GN] * GH + h];
                if ((j >= nn) || ((i >= nn) && (j >= 1))) v = -1.0e9f;
                sS[iloc * 520 + j] = v;
            }
        }
    }
    __syncthreads();

    // ---- phase 2: softmax (8 threads per row) --------------------------
    const int row = tid >> 3, seg = tid & 7;
    {
        int base = row * 520 + seg * 64;
        float m = -3.0e38f;
        for (int c = 0; c < 64; ++c) m = fmaxf(m, sS[base + c]);
        sRed[row * 9 + seg] = m;
        __syncthreads();
        if (tid < 16) {
            float mm = sRed[tid * 9];
            for (int s = 1; s < 8; ++s) mm = fmaxf(mm, sRed[tid * 9 + s]);
            sRed[tid * 9 + 8] = mm;
        }
        __syncthreads();
        float rm = sRed[row * 9 + 8];
        float sum = 0.0f;
        for (int c = 0; c < 64; ++c) {
            float e = __expf(sS[base + c] - rm);
            sS[base + c] = e;
            sum += e;
        }
        sRed[row * 9 + seg] = sum;
        __syncthreads();
        if (tid < 16) {
            float ss = 0.0f;
            for (int s = 0; s < 8; ++s) ss += sRed[tid * 9 + s];
            sRed[tid * 9 + 8] = 1.0f / ss;   // deferred normalization
        }
        __syncthreads();
    }

    // ---- phase 3: O = P @ V --------------------------------------------
    f32x8 oacc[2];
    oacc[0] = zero; oacc[1] = zero;
    for (int sc4 = 0; sc4 < 4; ++sc4) {
        int kj = sc4 * 128;
        __syncthreads();
        {   // stage V chunk transposed: sVt[d][key]
            const u32x4* vp =
                (const u32x4*)(vb + ((size_t)(b * GN + kj + tid)) * GF + h * 32);
            union { u32x4 q[4]; ushort_t u[32]; } tv;
#pragma unroll
            for (int c4 = 0; c4 < 4; ++c4) tv.q[c4] = vp[c4];
#pragma unroll
            for (int d = 0; d < 32; ++d) sVt[d * 136 + tid] = tv.u[d];
        }
        __syncthreads();
        FragB16 pf;
        int colbase = kj + wv * 32;
        int kA = hi ? 8 : 0, kB = hi ? 24 : 16;
#pragma unroll
        for (int e = 0; e < 8; ++e) {
            pf.u[e]     = f2bf(sS[lrow * 520 + colbase + kA + e]);
            pf.u[8 + e] = f2bf(sS[lrow * 520 + colbase + kB + e]);
        }
#pragma unroll
        for (int nt = 0; nt < 2; ++nt) {
            int d = nt * 16 + lrow;
            FragB16 vf;
            int ob3 = hi ? 8 : 0;
            vf.q[0] = *(const u32x4*)&sVt[d * 136 + wv * 32 + ob3];
            vf.q[1] = *(const u32x4*)&sVt[d * 136 + wv * 32 + ob3 + 16];
            oacc[nt] = __builtin_amdgcn_wmma_f32_16x16x32_bf16(
                false, pf.v, false, vf.v, (short)0, oacc[nt], false, false);
        }
    }
#pragma unroll
    for (int nt = 0; nt < 2; ++nt)
#pragma unroll
        for (int r = 0; r < 8; ++r)
            sO[wv * 512 + (r + (hi ? 8 : 0)) * 32 + nt * 16 + lrow] = oacc[nt][r];
    __syncthreads();
#pragma unroll
    for (int e = 0; e < 4; ++e) {
        int idx  = tid + e * 128;
        int iloc = idx >> 5, d = idx & 31;
        float s = sO[iloc * 32 + d] + sO[512 + iloc * 32 + d] +
                  sO[1024 + iloc * 32 + d] + sO[1536 + iloc * 32 + d];
        s *= sRed[iloc * 9 + 8];
        ob[((size_t)(b * GN + qi + iloc)) * GF + h * 32 + d] = f2bf(s);
    }
}

// ---------------------------------------------------------------------------
// Residual add + LayerNorm; writes f32 and bf16 copies. One block per row.
// ---------------------------------------------------------------------------
__global__ __launch_bounds__(256) void add_ln_kernel(
    const float* xin, const float* __restrict__ dres,
    const float* __restrict__ gs, const float* __restrict__ gb, float* xo,
    ushort_t* __restrict__ xbo) {
    __shared__ float red[256];
    const int row = blockIdx.x, tid = threadIdx.x;
    const size_t base = (size_t)row * GF;
    float v0 = xin[base + tid] + dres[base + tid];
    float v1 = xin[base + tid + 256] + dres[base + tid + 256];
    red[tid] = v0 + v1;
    __syncthreads();
    for (int s = 128; s > 0; s >>= 1) {
        if (tid < s) red[tid] += red[tid + s];
        __syncthreads();
    }
    float mean = red[0] * (1.0f / GF);
    __syncthreads();
    float d0 = v0 - mean, d1 = v1 - mean;
    red[tid] = d0 * d0 + d1 * d1;
    __syncthreads();
    for (int s = 128; s > 0; s >>= 1) {
        if (tid < s) red[tid] += red[tid + s];
        __syncthreads();
    }
    float rstd = rsqrtf(red[0] * (1.0f / GF) + 1e-5f);
    float o0 = d0 * rstd * gs[tid] + gb[tid];
    float o1 = d1 * rstd * gs[tid + 256] + gb[tid + 256];
    xo[base + tid] = o0;
    xo[base + tid + 256] = o1;
    xbo[base + tid] = f2bf(o0);
    xbo[base + tid + 256] = f2bf(o1);
}

// ---------------------------------------------------------------------------
// Masked mean-pool over valid nodes + linear classifier + sigmoid.
// ---------------------------------------------------------------------------
__global__ __launch_bounds__(256) void pool_kernel(
    const float* __restrict__ x, const int* __restrict__ num_nodes,
    const float* __restrict__ cw, const float* __restrict__ cb,
    float* __restrict__ out) {
    __shared__ float red[256];
    const int b = blockIdx.x, tid = threadIdx.x;
    const int nn = num_nodes[b];
    float p = 0.0f;
    for (int n = 0; n < nn; ++n) {
        const float* xr = x + ((size_t)(b * GN + n)) * GF;
        for (int f = tid; f < GF; f += 256) p += xr[f] * cw[f];
    }
    red[tid] = p;
    __syncthreads();
    for (int s = 128; s > 0; s >>= 1) {
        if (tid < s) red[tid] += red[tid + s];
        __syncthreads();
    }
    if (tid == 0) {
        float z = red[0] / (float)nn + cb[0];
        out[b] = 1.0f / (1.0f + __expf(-z));
    }
}

// ---------------------------------------------------------------------------
// Host orchestration
// ---------------------------------------------------------------------------
extern "C" void kernel_launch(void* const* d_in, const int* in_sizes, int n_in,
                              void* d_out, int out_size, void* d_ws,
                              size_t ws_size, hipStream_t stream) {
    (void)in_sizes; (void)n_in; (void)out_size; (void)ws_size;

    const float* nfeats   = (const float*)d_in[0];
    const int*   degrees  = (const int*)d_in[1];
    const int*   dist_idx = (const int*)d_in[2];
    const int*   num_node = (const int*)d_in[3];
    const float* din      = (const float*)d_in[4];
    const float* dou      = (const float*)d_in[5];
    const float* spatial  = (const float*)d_in[6];
    const float* Wq  = (const float*)d_in[7];
    const float* bq  = (const float*)d_in[8];
    const float* Wk  = (const float*)d_in[9];
    const float* bk  = (const float*)d_in[10];
    const float* Wv  = (const float*)d_in[11];
    const float* bv  = (const float*)d_in[12];
    const float* Wo  = (const float*)d_in[13];
    const float* bo  = (const float*)d_in[14];
    const float* l1s = (const float*)d_in[15];
    const float* l1b = (const float*)d_in[16];
    const float* W1  = (const float*)d_in[17];
    const float* b1  = (const float*)d_in[18];
    const float* W2  = (const float*)d_in[19];
    const float* b2  = (const float*)d_in[20];
    const float* l2s = (const float*)d_in[21];
    const float* l2b = (const float*)d_in[22];
    const float* cw  = (const float*)d_in[23];
    const float* cbv = (const float*)d_in[24];

    char* ws = (char*)d_ws;
    size_t off = 0;
    auto alloc = [&](size_t bytes) -> void* {
        void* p = ws + off;
        off = (off + bytes + 255) & ~(size_t)255;
        return p;
    };
    ushort_t* wtq = (ushort_t*)alloc((size_t)GL * GF * GF * 2);
    ushort_t* wtk = (ushort_t*)alloc((size_t)GL * GF * GF * 2);
    ushort_t* wtv = (ushort_t*)alloc((size_t)GL * GF * GF * 2);
    ushort_t* wto = (ushort_t*)alloc((size_t)GL * GF * GF * 2);
    ushort_t* wt1 = (ushort_t*)alloc((size_t)GL * GHID * GF * 2);
    ushort_t* wt2 = (ushort_t*)alloc((size_t)GL * GF * GHID * 2);
    float*    xf  = (float*)alloc((size_t)GM * GF * 4);
    ushort_t* xb  = (ushort_t*)alloc((size_t)GM * GF * 2);
    ushort_t* qbf = (ushort_t*)alloc((size_t)GM * GF * 2);
    ushort_t* kbf = (ushort_t*)alloc((size_t)GM * GF * 2);
    ushort_t* vbf = (ushort_t*)alloc((size_t)GM * GF * 2);
    ushort_t* obf = (ushort_t*)alloc((size_t)GM * GF * 2);
    float*    tmp = (float*)alloc((size_t)GM * GF * 4);
    ushort_t* hbf = (ushort_t*)alloc((size_t)GM * GHID * 2);

    // ---- weight prep (convert + transpose to bf16) ----------------------
    for (int l = 0; l < GL; ++l) {
        size_t oF = (size_t)l * GF * GF;
        size_t oH = (size_t)l * GF * GHID;
        convt_kernel<<<(GF * GF) / 256, 256, 0, stream>>>(Wq + oF, wtq + oF, GF, GF);
        convt_kernel<<<(GF * GF) / 256, 256, 0, stream>>>(Wk + oF, wtk + oF, GF, GF);
        convt_kernel<<<(GF * GF) / 256, 256, 0, stream>>>(Wv + oF, wtv + oF, GF, GF);
        convt_kernel<<<(GF * GF) / 256, 256, 0, stream>>>(Wo + oF, wto + oF, GF, GF);
        convt_kernel<<<(GF * GHID) / 256, 256, 0, stream>>>(W1 + oH, wt1 + oH, GF, GHID);
        convt_kernel<<<(GHID * GF) / 256, 256, 0, stream>>>(W2 + oH, wt2 + oH, GHID, GF);
    }

    embed_kernel<<<(size_t)GM * GF / 256, 256, 0, stream>>>(
        nfeats, degrees, num_node, din, dou, xf, xb);

    const float  scale = 0.17677669529663687f;  // 32^-0.5
    const dim3   gridF(GF / 64, GM / 128);
    const dim3   gridH(GHID / 64, GM / 128);
    const size_t gemmLds = 30720;   // double-buffered A/B tiles
    const size_t attnLds = 62272;

    for (int l = 0; l < GL; ++l) {
        size_t oF = (size_t)l * GF * GF;
        size_t oH = (size_t)l * GF * GHID;
        // Q (pre-scaled), K, V projections -> bf16
        gemm_bf16_kernel<<<gridF, 256, gemmLds, stream>>>(
            xb, wtq + oF, bq + l * GF, nullptr, qbf, GM, GF, GF, scale, 0);
        gemm_bf16_kernel<<<gridF, 256, gemmLds, stream>>>(
            xb, wtk + oF, bk + l * GF, nullptr, kbf, GM, GF, GF, 1.0f, 0);
        gemm_bf16_kernel<<<gridF, 256, gemmLds, stream>>>(
            xb, wtv + oF, bv + l * GF, nullptr, vbf, GM, GF, GF, 1.0f, 0);
        // attention
        attn_kernel<<<GB * GH * (GN / 16), 128, attnLds, stream>>>(
            qbf, kbf, vbf, dist_idx, spatial, num_node, obf);
        // output projection -> f32
        gemm_bf16_kernel<<<gridF, 256, gemmLds, stream>>>(
            obf, wto + oF, bo + l * GF, tmp, nullptr, GM, GF, GF, 1.0f, 0);
        add_ln_kernel<<<GM, 256, 0, stream>>>(xf, tmp, l1s + l * GF,
                                              l1b + l * GF, xf, xb);
        // FFN
        gemm_bf16_kernel<<<gridH, 256, gemmLds, stream>>>(
            xb, wt1 + oH, b1 + l * GHID, nullptr, hbf, GM, GF, GHID, 1.0f, 1);
        gemm_bf16_kernel<<<gridF, 256, gemmLds, stream>>>(
            hbf, wt2 + oH, b2 + l * GF, tmp, nullptr, GM, GHID, GF, 1.0f, 0);
        add_ln_kernel<<<GM, 256, 0, stream>>>(xf, tmp, l2s + l * GF,
                                              l2b + l * GF, xf, xb);
    }

    pool_kernel<<<GB, 256, 0, stream>>>(xf, num_node, cw, cbv, (float*)d_out);
}
